// GatingNetwork_20968030339721
// MI455X (gfx1250) — matrix operand order repaired and checked by reading, not compiled, not run on hardware
//
#include <hip/hip_runtime.h>
#include <hip/hip_bf16.h>

typedef __attribute__((ext_vector_type(16))) __bf16 v16bf;
typedef __attribute__((ext_vector_type(8)))  float  v8f;

#define BB 131072
#define SS 4
#define DD 512
#define EE 64
#define ROWS_PER_BLOCK 128   // 8 waves * 16 rows
#define NEG_HUGE (-3.402823466e38f)

// ---------------------------------------------------------------------------
// Kernel 1: pack w_gate (E x D f32) into bf16 hi/lo planes; zero accumulators.
// hi = bf16(w); lo = bf16(w - f32(hi))  -> bf16x3 GEMM recovers ~fp32 accuracy.
// ---------------------------------------------------------------------------
__global__ void __launch_bounds__(256) moe_prep_kernel(
    const float* __restrict__ w_gate,
    unsigned short* __restrict__ w_hi_u,
    unsigned short* __restrict__ w_lo_u,
    float* __restrict__ g_accum /* 128 floats: [0..63]=importance, [64..127]=load */)
{
    int i = blockIdx.x * blockDim.x + threadIdx.x;
    __bf16* w_hi = (__bf16*)w_hi_u;
    __bf16* w_lo = (__bf16*)w_lo_u;
    if (i < EE * DD) {
        float w = w_gate[i];
        __bf16 hi = (__bf16)w;
        __bf16 lo = (__bf16)(w - (float)hi);
        w_hi[i] = hi;
        w_lo[i] = lo;
    }
    if (i < 2 * EE) g_accum[i] = 0.0f;   // re-zeroed every launch (replay-safe)
}

// ---------------------------------------------------------------------------
// Kernel 2: gating. Each wave computes a 16x64 logits tile via
// v_wmma_f32_16x16x32_bf16 (bf16x3 split), then block does top-2 softmax,
// builds the sparse gates tile in LDS and stores it coalesced.
// ---------------------------------------------------------------------------
__global__ void __launch_bounds__(256) moe_gating_kernel(
    const float* __restrict__ hs,            // (B, S, D) f32; we read s==0 slice
    const unsigned short* __restrict__ w_hi_u,
    const unsigned short* __restrict__ w_lo_u,
    float* __restrict__ gates_out,           // (B, E) f32
    float* __restrict__ g_accum)             // importance[64] | load[64]
{
    __shared__ float lds[ROWS_PER_BLOCK * 65 + 2 * EE];
    float* imp_s  = lds + ROWS_PER_BLOCK * 65;
    float* load_s = imp_s + EE;

    const __bf16* w_hi = (const __bf16*)w_hi_u;
    const __bf16* w_lo = (const __bf16*)w_lo_u;

    const int tid  = threadIdx.x;
    const int wave = tid >> 5;
    const int lane = tid & 31;
    const int hh   = lane >> 4;      // lane half (0/1)
    const int l16  = lane & 15;
    const int rowbase = blockIdx.x * ROWS_PER_BLOCK + wave * 16;

    // A-fragment source row for this lane (M = l16)
    const float* xrow = hs + (size_t)(rowbase + l16) * (SS * DD);

    v8f acc[4];
#pragma unroll
    for (int t = 0; t < 4; ++t)
#pragma unroll
        for (int v = 0; v < 8; ++v) acc[t][v] = 0.0f;

#pragma unroll 1
    for (int s = 0; s < DD / 32; ++s) {
        const int kbase = s * 32;
        // A (16x32 bf16) per-lane data: lanes 0-15 -> K 0..15,
        // lanes 16-31 -> K 16..31 (contiguous, matching B).
        const float* p0 = xrow + kbase + hh * 16;
        const float* p1 = xrow + kbase + hh * 16 + 8;
        float4 c0a = *(const float4*)(p0);
        float4 c0b = *(const float4*)(p0 + 4);
        float4 c1a = *(const float4*)(p1);
        float4 c1b = *(const float4*)(p1 + 4);
        float av[16] = { c0a.x, c0a.y, c0a.z, c0a.w,  c0b.x, c0b.y, c0b.z, c0b.w,
                         c1a.x, c1a.y, c1a.z, c1a.w,  c1b.x, c1b.y, c1b.z, c1b.w };

        if (s + 1 < DD / 32) __builtin_prefetch(xrow + kbase + 32, 0, 1);

        v16bf a_hi, a_lo;
#pragma unroll
        for (int i = 0; i < 16; ++i) {
            float v = av[i];
            __bf16 h = (__bf16)v;
            a_hi[i] = h;
            a_lo[i] = (__bf16)(v - (float)h);
        }

#pragma unroll
        for (int t = 0; t < 4; ++t) {
            // B (32x16 bf16): lane -> expert e = t*16 + l16; each lane holds 16
            // consecutive K values (lanes 0-15: K 0..15, lanes 16-31: K 16..31)
            // -> contiguous 32B load from the packed bf16 W planes.
            const int e = t * 16 + l16;
            v16bf b_hi = *(const v16bf*)(w_hi + (size_t)e * DD + kbase + hh * 16);
            v16bf b_lo = *(const v16bf*)(w_lo + (size_t)e * DD + kbase + hh * 16);
            acc[t] = __builtin_amdgcn_wmma_f32_16x16x32_bf16(
                false, a_lo, false, b_hi, (short)0, acc[t], false, false);
            acc[t] = __builtin_amdgcn_wmma_f32_16x16x32_bf16(
                false, a_hi, false, b_lo, (short)0, acc[t], false, false);
            acc[t] = __builtin_amdgcn_wmma_f32_16x16x32_bf16(
                false, a_hi, false, b_hi, (short)0, acc[t], false, false);
        }
    }

    // C layout: VGPR v of lane -> M = v + 8*hh, N = l16 (per 16x16 f32 D layout)
#pragma unroll
    for (int t = 0; t < 4; ++t)
#pragma unroll
        for (int v = 0; v < 8; ++v) {
            int r = wave * 16 + v + 8 * hh;
            int c = t * 16 + l16;
            lds[r * 65 + c] = acc[t][v];
        }
    __syncthreads();

    // top-2 + softmax per row (threads 0..127, one per row)
    float g1 = 0.0f, g2 = 0.0f;
    int   e1 = 0, e2 = 0;
    if (tid < ROWS_PER_BLOCK) {
        const float* lr = lds + tid * 65;
        float v1 = NEG_HUGE, v2 = NEG_HUGE;
        int   i1 = 0, i2 = 0;
        for (int e = 0; e < EE; ++e) {
            float v = lr[e];
            if (v > v1)      { v2 = v1; i2 = i1; v1 = v; i1 = e; }
            else if (v > v2) { v2 = v;  i2 = e; }
        }
        float x1 = __expf(0.0f);          // v1 - v1
        float x2 = __expf(v2 - v1);
        float inv = 1.0f / (x1 + x2);
        g1 = x1 * inv; g2 = x2 * inv; e1 = i1; e2 = i2;
    }
    __syncthreads();

    // zero gates tile + per-block expert accumulators
    for (int i = tid; i < ROWS_PER_BLOCK * 65 + 2 * EE; i += blockDim.x) lds[i] = 0.0f;
    __syncthreads();

    if (tid < ROWS_PER_BLOCK) {
        lds[tid * 65 + e1] = g1;
        lds[tid * 65 + e2] = g2;
        atomicAdd(&imp_s[e1], g1);
        atomicAdd(&imp_s[e2], g2);
        atomicAdd(&load_s[e1], 1.0f);
        atomicAdd(&load_s[e2], 1.0f);
    }
    __syncthreads();

    // coalesced 128x64 gates tile store
    float* gout = gates_out + (size_t)blockIdx.x * ROWS_PER_BLOCK * EE;
    for (int i = tid; i < ROWS_PER_BLOCK * EE; i += blockDim.x) {
        int r = i >> 6, c = i & 63;
        gout[i] = lds[r * 65 + c];
    }

    // one global atomic per expert per block
    if (tid < EE) {
        atomicAdd(&g_accum[tid], imp_s[tid]);
        atomicAdd(&g_accum[EE + tid], load_s[tid]);
    }
}

// ---------------------------------------------------------------------------
// Kernel 3: loss = 0.01 * (cv2(importance) + cv2(load)), cv2 = var(ddof=1)/mean^2
// ---------------------------------------------------------------------------
__global__ void moe_loss_kernel(const float* __restrict__ g_accum,
                                float* __restrict__ out_loss)
{
    if (threadIdx.x == 0 && blockIdx.x == 0) {
        float cv[2];
        for (int a = 0; a < 2; ++a) {
            const float* p = g_accum + a * EE;
            float s = 0.0f, s2 = 0.0f;
            for (int e = 0; e < EE; ++e) { float v = p[e]; s += v; s2 += v * v; }
            float mean = s / (float)EE;
            float var  = (s2 - (float)EE * mean * mean) / (float)(EE - 1);
            cv[a] = var / (mean * mean + 1e-10f);
        }
        out_loss[0] = (cv[0] + cv[1]) * 0.01f;
    }
}

// ---------------------------------------------------------------------------
extern "C" void kernel_launch(void* const* d_in, const int* in_sizes, int n_in,
                              void* d_out, int out_size, void* d_ws, size_t ws_size,
                              hipStream_t stream)
{
    const float* hs     = (const float*)d_in[0];   // (B,S,D) f32
    const float* w_gate = (const float*)d_in[1];   // (E,D)   f32

    float* gates = (float*)d_out;                  // (B,E) f32
    float* loss  = (float*)d_out + (size_t)BB * EE;

    // workspace: w_hi (64KB bf16) | w_lo (64KB bf16) | g_accum (128 f32)
    unsigned short* w_hi = (unsigned short*)d_ws;
    unsigned short* w_lo = w_hi + EE * DD;
    float* g_accum = (float*)((char*)d_ws + (size_t)2 * EE * DD * sizeof(unsigned short));

    moe_prep_kernel<<<(EE * DD + 255) / 256, 256, 0, stream>>>(w_gate, w_hi, w_lo, g_accum);
    moe_gating_kernel<<<BB / ROWS_PER_BLOCK, 256, 0, stream>>>(hs, w_hi, w_lo, gates, g_accum);
    moe_loss_kernel<<<1, 64, 0, stream>>>(g_accum, loss);
}
